// ExtractModel_79173427134503
// MI455X (gfx1250) — compile-verified, matrix operand order checked by reading, not run on hardware
//
#include <hip/hip_runtime.h>
#include <math.h>

// ---- problem constants (match reference) ----
#define B_   32
#define LS_  64
#define LE_  16
#define NV_  32768            // B*LS*LE
#define V_   6000
#define D_   256
#define NT_  (V_ / 16)        // 375 vocab tiles of 16
#define INV_D (1.0f / 256.0f)
#define INV_T 10.0f           // 1/temperature

typedef _Float16 half8_t __attribute__((ext_vector_type(8)));
typedef _Float16 v16h    __attribute__((ext_vector_type(16)));
typedef float    v8f     __attribute__((ext_vector_type(8)));

// -----------------------------------------------------------------------------
// Kernel 1: split vocab_repr (fp32) into f16 hi/lo pair and compute ||v||^2.
// grid = V_, block = 64 (each thread handles 4 contiguous elements).
// -----------------------------------------------------------------------------
__global__ __launch_bounds__(64) void prep_vocab(
    const float* __restrict__ vocab,
    _Float16* __restrict__ vhi, _Float16* __restrict__ vlo,
    float* __restrict__ v2)
{
    const int v   = blockIdx.x;
    const int tid = threadIdx.x;
    const int base = v * D_ + tid * 4;

    float4 x = *(const float4*)(vocab + base);
    float xs[4] = {x.x, x.y, x.z, x.w};
    float ssq = 0.0f;
#pragma unroll
    for (int j = 0; j < 4; ++j) {
        float f = xs[j];
        _Float16 h = (_Float16)f;
        vhi[base + j] = h;
        vlo[base + j] = (_Float16)(f - (float)h);
        ssq += f * f;
    }
    __shared__ float red[64];
    red[tid] = ssq;
    __syncthreads();
#pragma unroll
    for (int s = 32; s > 0; s >>= 1) {
        if (tid < s) red[tid] += red[tid + s];
        __syncthreads();
    }
    if (tid == 0) v2[v] = red[0];
}

// -----------------------------------------------------------------------------
// Kernel 2: fused split-f16 WMMA GEMM + streaming soft-min over vocab.
// grid = NV_/64 blocks, block = 128 (4 waves, 16 candidate rows per wave).
// Each 16x16 tile accumulates hi*hi + hi*lo + lo*hi (3 WMMAs per K=32 step).
// -----------------------------------------------------------------------------
__global__ __launch_bounds__(128) void gemm_softmin(
    const float* __restrict__ cand,
    const _Float16* __restrict__ vhi, const _Float16* __restrict__ vlo,
    const float* __restrict__ v2,
    const int* __restrict__ bi, const int* __restrict__ lsi,
    const int* __restrict__ lei, const int* __restrict__ vlen,
    float* __restrict__ out, int* __restrict__ mvocab)
{
    const int lane    = threadIdx.x & 31;
    const int wave    = threadIdx.x >> 5;
    const int rowBase = blockIdx.x * 64 + wave * 16;
    const int mrow    = lane & 15;          // A row / B column handled by this lane
    const int khalf   = (lane >> 4) * 8;    // K sub-offset for 16-bit WMMA layout

    // ---- preload A fragments (16 rows x 256 K) as f16 hi/lo, fp32 row norms ----
    v16h ahi[8], alo[8];
    float ssq = 0.0f;
#pragma unroll
    for (int kc = 0; kc < 8; ++kc) {
        const float* src = cand + (size_t)(rowBase + mrow) * D_ + kc * 32 + khalf;
        float4 c0 = *(const float4*)(src);
        float4 c1 = *(const float4*)(src + 4);
        float4 c2 = *(const float4*)(src + 16);
        float4 c3 = *(const float4*)(src + 20);
        float xs[16] = {c0.x,c0.y,c0.z,c0.w, c1.x,c1.y,c1.z,c1.w,
                        c2.x,c2.y,c2.z,c2.w, c3.x,c3.y,c3.z,c3.w};
#pragma unroll
        for (int j = 0; j < 16; ++j) {
            float f = xs[j];
            _Float16 h = (_Float16)f;
            ahi[kc][j] = h;
            alo[kc][j] = (_Float16)(f - (float)h);
            ssq += f * f;
        }
    }
    // full ||c||^2 of row (lane&15): combine the two K-halves
    float c2row = ssq + __shfl_xor(ssq, 16, 32);

    // ---- streaming soft-min state (per lane, per accumulator row r) ----
    float rmin[8], ssum[8], wsum[8];
    int   idx[8];
#pragma unroll
    for (int r = 0; r < 8; ++r) { rmin[r] = 3.0e38f; ssum[r] = 0.0f; wsum[r] = 0.0f; idx[r] = 0; }

    for (int t = 0; t < NT_; ++t) {
        const int vrow = t * 16 + mrow;                 // vocab index for this lane's column
        const float v2s = v2[vrow] * INV_D;

        v8f acc = {0.f,0.f,0.f,0.f,0.f,0.f,0.f,0.f};
#pragma unroll
        for (int kc = 0; kc < 8; ++kc) {
            const _Float16* bh = vhi + (size_t)vrow * D_ + kc * 32 + khalf;
            const _Float16* bl = vlo + (size_t)vrow * D_ + kc * 32 + khalf;
            half8_t h0 = *(const half8_t*)(bh);
            half8_t h1 = *(const half8_t*)(bh + 16);
            half8_t l0 = *(const half8_t*)(bl);
            half8_t l1 = *(const half8_t*)(bl + 16);
            v16h bH = __builtin_shufflevector(h0, h1, 0,1,2,3,4,5,6,7,8,9,10,11,12,13,14,15);
            v16h bL = __builtin_shufflevector(l0, l1, 0,1,2,3,4,5,6,7,8,9,10,11,12,13,14,15);
            // D = A*B + C  (fp32 accumulate): hi*hi + hi*lo + lo*hi
            acc = __builtin_amdgcn_wmma_f32_16x16x32_f16(false, ahi[kc], false, bH, (short)0, acc, false, false);
            acc = __builtin_amdgcn_wmma_f32_16x16x32_f16(false, ahi[kc], false, bL, (short)0, acc, false, false);
            acc = __builtin_amdgcn_wmma_f32_16x16x32_f16(false, alo[kc], false, bH, (short)0, acc, false, false);
        }
        // u = (||v||^2 - 2*dot)/D ; ||c||^2/D shift cancels in softmax weights
#pragma unroll
        for (int r = 0; r < 8; ++r) {
            float u = fmaf(acc[r], -2.0f * INV_D, v2s);
            float e = __expf(u * -INV_T);
            bool  c = u < rmin[r];
            rmin[r] = c ? u : rmin[r];
            idx[r]  = c ? vrow : idx[r];
            ssum[r] += e;
            wsum[r] = fmaf(u, e, wsum[r]);
        }
    }

    // ---- reduce soft-min state across the 16 lanes of each half-wave ----
#pragma unroll
    for (int r = 0; r < 8; ++r) {
#pragma unroll
        for (int m = 1; m <= 8; m <<= 1) {
            float os = __shfl_xor(ssum[r], m, 32);
            float ow = __shfl_xor(wsum[r], m, 32);
            float om = __shfl_xor(rmin[r], m, 32);
            int   oi = __shfl_xor(idx[r],  m, 32);
            ssum[r] += os;
            wsum[r] += ow;
            bool take = (om < rmin[r]) || (om == rmin[r] && oi < idx[r]);
            rmin[r] = take ? om : rmin[r];
            idx[r]  = take ? oi : idx[r];
        }
    }
    // row M handled by (lane 0 -> M=r) and (lane 16 -> M=8+r); fetch ||c||^2
    // for those rows while all lanes are still active (permute needs EXEC).
    float c2m[8];
#pragma unroll
    for (int r = 0; r < 8; ++r) c2m[r] = __shfl(c2row, r + khalf, 32);

    if ((lane & 15) == 0) {
#pragma unroll
        for (int r = 0; r < 8; ++r) {
            float med  = c2m[r] * INV_D + wsum[r] / ssum[r];      // matched_ed_dist
            float y    = 1.0f - 40.0f * med;                       // 1 - 2x/0.05
            float celu = (y > 0.0f) ? y : expm1f(y);
            float score = (celu + 1.0f) * 0.5f * (float)vlen[idx[r]];
            int g  = rowBase + r + khalf;
            int di = bi[g] * (LS_ * LE_) + lsi[g] * LE_ + lei[g];
            out[4 * B_ + di] = score;      // dense_score output region
            mvocab[di]       = idx[r];     // dense_vocab (workspace)
        }
    }
}

// -----------------------------------------------------------------------------
// Kernel 3: per-batch soft-max over 1024 dense scores (max-stabilized),
// argmax -> start/end/vocab. grid = B_, block = 128 (8 elements/thread).
// -----------------------------------------------------------------------------
__global__ __launch_bounds__(128) void softmax_best(
    const float* __restrict__ dense, const int* __restrict__ mvocab,
    float* __restrict__ out)
{
    __shared__ float rv[128];
    __shared__ int   ri[128];
    const int b   = blockIdx.x;
    const int tid = threadIdx.x;
    const float* row = dense + b * (LS_ * LE_);

    float x[8];
#pragma unroll
    for (int j = 0; j < 8; ++j) x[j] = row[tid + j * 128];

    float m = -3.0e38f; int mi = 0;
#pragma unroll
    for (int j = 0; j < 8; ++j) {
        int e = tid + j * 128;
        if (x[j] > m) { m = x[j]; mi = e; }
    }
    rv[tid] = m; ri[tid] = mi;
    __syncthreads();
#pragma unroll
    for (int s = 64; s > 0; s >>= 1) {
        if (tid < s) {
            bool take = (rv[tid + s] > rv[tid]) ||
                        (rv[tid + s] == rv[tid] && ri[tid + s] < ri[tid]);
            if (take) { rv[tid] = rv[tid + s]; ri[tid] = ri[tid + s]; }
        }
        __syncthreads();
    }
    const float Mx  = rv[0];
    const int   bix = ri[0];
    __syncthreads();

    float s = 0.0f, w = 0.0f;
#pragma unroll
    for (int j = 0; j < 8; ++j) {
        float e = __expf((x[j] - Mx) * INV_T);
        s += e;
        w = fmaf(x[j], e, w);
    }
    rv[tid] = s;
    __syncthreads();
#pragma unroll
    for (int t = 64; t > 0; t >>= 1) { if (tid < t) rv[tid] += rv[tid + t]; __syncthreads(); }
    const float S = rv[0];
    __syncthreads();
    rv[tid] = w;
    __syncthreads();
#pragma unroll
    for (int t = 64; t > 0; t >>= 1) { if (tid < t) rv[tid] += rv[tid + t]; __syncthreads(); }
    const float W = rv[0];

    if (tid == 0) {
        int start = bix >> 4;
        int endp  = start + (bix & 15);
        int voc   = mvocab[b * (LS_ * LE_) + bix];
        out[b]          = W / S;          // best_matched_score
        out[B_ + b]     = (float)start;   // start
        out[2 * B_ + b] = (float)endp;    // end
        out[3 * B_ + b] = (float)voc;     // best_matched_vocab
    }
}

// -----------------------------------------------------------------------------
extern "C" void kernel_launch(void* const* d_in, const int* in_sizes, int n_in,
                              void* d_out, int out_size, void* d_ws, size_t ws_size,
                              hipStream_t stream)
{
    const float* cand  = (const float*)d_in[0];   // [NV, D] f32
    const float* vocab = (const float*)d_in[1];   // [V, D]  f32
    const int*   bi    = (const int*)d_in[2];
    const int*   lsi   = (const int*)d_in[3];
    const int*   lei   = (const int*)d_in[4];
    const int*   vlen  = (const int*)d_in[5];
    float*       out   = (float*)d_out;

    char* ws = (char*)d_ws;
    _Float16* vhi   = (_Float16*)ws;                                  // V*D f16
    _Float16* vlo   = (_Float16*)(ws + (size_t)V_ * D_ * 2);          // V*D f16
    float*    v2    = (float*)   (ws + (size_t)V_ * D_ * 4);          // V f32
    int*      mvoc  = (int*)     (ws + (size_t)V_ * D_ * 4 + V_ * 4); // NV i32

    prep_vocab  <<<V_,        64, 0, stream>>>(vocab, vhi, vlo, v2);
    gemm_softmin<<<NV_ / 64, 128, 0, stream>>>(cand, vhi, vlo, v2,
                                               bi, lsi, lei, vlen, out, mvoc);
    softmax_best<<<B_,       128, 0, stream>>>(out + 4 * B_, mvoc, out);
}
